// PDECNN1_8839042695229
// MI455X (gfx1250) — matrix operand order repaired and checked by reading, not compiled, not run on hardware
//
#include <hip/hip_runtime.h>
#include <hip/hip_bf16.h>
#include <math.h>

// ---------------------------------------------------------------------------
// PDECNN pipeline for MI455X (gfx1250, wave32, WMMA).
// Heavy contractions -> v_wmma_f32_16x16x32_bf16 (bf16 in, f32 acc).
// B operands pre-converted to bf16 (K padded to 1216) and staged to LDS with
// global_load_async_to_lds_b128 (ASYNCcnt path).
// ---------------------------------------------------------------------------

typedef __attribute__((ext_vector_type(16))) __bf16 v16bf;
typedef __attribute__((ext_vector_type(8)))  float  v8f;
typedef int vi4 __attribute__((vector_size(16)));

union Frag16 { v16bf v; unsigned int u[8]; };

__device__ __forceinline__ unsigned short f2bf(float f) {
    union { float f; unsigned int u; } x; x.f = f;
    unsigned int r = x.u + 0x7FFFu + ((x.u >> 16) & 1u);   // round-to-nearest-even
    return (unsigned short)(r >> 16);
}
__device__ __forceinline__ unsigned int pk2bf(float a, float b) {
    return (unsigned int)f2bf(a) | ((unsigned int)f2bf(b) << 16);
}

#if __has_builtin(__builtin_amdgcn_global_load_async_to_lds_b128)
#define HAVE_ASYNC_LDS 1
#else
#define HAVE_ASYNC_LDS 0
#endif

__device__ __forceinline__ void async_copy16(void* lds, const void* g) {
#if HAVE_ASYNC_LDS
    __builtin_amdgcn_global_load_async_to_lds_b128(
        (__attribute__((address_space(1))) vi4*)g,
        (__attribute__((address_space(3))) vi4*)lds, 0, 0);
#else
    *(uint4*)lds = *(const uint4*)g;
#endif
}
__device__ __forceinline__ void wait_async0() {
#if HAVE_ASYNC_LDS
#if __has_builtin(__builtin_amdgcn_s_wait_asynccnt)
    __builtin_amdgcn_s_wait_asynccnt(0);
#else
    asm volatile("s_wait_asynccnt 0x0" ::: "memory");
#endif
#endif
}

#define TM   128
#define TN   80
#define TK   32
#define KPAD 1216   // 1200 padded to a multiple of 32 (zero tail)

// ---------------------------------------------------------------------------
// bf16 matrix pre-conversion with zero K-padding: src (rows x K) f32 ->
// dst (rows x KPAD) bf16.
// ---------------------------------------------------------------------------
__global__ void cvt_mat_bf16_k(const float* __restrict__ src,
                               unsigned short* __restrict__ dst, int rows, int K)
{
    int idx = blockIdx.x * 256 + threadIdx.x;
    int tot = rows * KPAD;
    if (idx >= tot) return;
    int r = idx / KPAD, k = idx % KPAD;
    float v = (k < K) ? src[(size_t)r * K + k] : 0.0f;
    dst[idx] = f2bf(v);
}

// ---------------------------------------------------------------------------
// Generic GEMM: D[b,l] = epilogue( sum_k M[l,k] * A'[b,k] ).
// A' optionally row-normalized during staging (templated, branch-free).
// Tile: 128 batch-rows x 80 L-cols (256 thr = 8 waves, 5 frags/wave).
// MODE: 0 plain | 1 (acc-mean)*rstd | 2 |acc| | 3 min(|acc|,aux1)
//       4 clamp(aux1-acc, aux2) w/ sign | 5 aux1-acc+aux2 | 6 acc*aux1[b]+aux2[b]
// ---------------------------------------------------------------------------
template <int ANORM, int MODE>
__global__ __launch_bounds__(256)
void gemm_bf16_ep(const float* __restrict__ A,
                  const unsigned short* __restrict__ Mb,   // bf16, rows x KPAD
                  float* __restrict__ D, int K, int L,
                  const float* __restrict__ meanp, const float* __restrict__ rstdp,
                  const float* __restrict__ aux1, const float* __restrict__ aux2)
{
    __shared__ unsigned short As[TM * TK];   // [row][k] bf16
    __shared__ unsigned short Bs[TN * TK];   // [l][k]  bf16
    const int tid  = threadIdx.x;
    const int wave = tid >> 5;
    const int lane = tid & 31;
    const int half = lane >> 4;
    const int l16  = lane & 15;
    const int m0   = blockIdx.y * TM;
    const int n0   = blockIdx.x * TN;

    v8f acc[5];
#pragma unroll
    for (int f = 0; f < 5; ++f)
#pragma unroll
        for (int i = 0; i < 8; ++i) acc[f][i] = 0.0f;

    const int ksteps = (K + TK - 1) / TK;
    for (int ks = 0; ks < ksteps; ++ks) {
        const int kb = ks * TK;

        // ---- B tile: pure 16B copies of pre-converted bf16 (async-to-LDS) ----
        for (int i = tid; i < TN * 4; i += 256) {       // 80 rows x 4 chunks
            int r = i >> 2, c = i & 3;
            const unsigned short* g = Mb + (size_t)(n0 + r) * KPAD + kb + c * 8;
            async_copy16(&Bs[r * TK + c * 8], (const void*)g);
        }
        // ---- A tile: float4 loads, fused normalization, pack to bf16 ----
        for (int i = tid; i < TM * 8; i += 256) {       // 128 rows x 8 chunks
            int r = i >> 3, c = i & 7;
            int gb = m0 + r, gk = kb + c * 4;
            float4 v = make_float4(0.f, 0.f, 0.f, 0.f);
            if (gk < K) v = *(const float4*)(A + (size_t)gb * K + gk);
            if (ANORM) {
                float mu = meanp[gb], rs = rstdp[gb];
                v.x = (v.x - mu) * rs; v.y = (v.y - mu) * rs;
                v.z = (v.z - mu) * rs; v.w = (v.w - mu) * rs;
            }
            unsigned int* d = (unsigned int*)&As[r * TK + c * 4];
            d[0] = pk2bf(v.x, v.y);
            d[1] = pk2bf(v.z, v.w);
        }
        wait_async0();
        __syncthreads();

        // A fragment (ISA 7.12.2 16x32 bf16 layout)
        Frag16 af;
        {
            const unsigned int* Arow =
                (const unsigned int*)&As[(wave * 16 + l16) * TK];
#pragma unroll
            for (int v = 0; v < 8; ++v) {
                int kbase = ((v & 4) ? 16 : 0) + half * 8 + (v & 3) * 2;
                af.u[v] = Arow[kbase >> 1];
            }
        }
        // preload all 5 B fragments, then issue the WMMAs back-to-back
        Frag16 bf_[5];
#pragma unroll
        for (int f = 0; f < 5; ++f) {
            const unsigned int* Brow =
                (const unsigned int*)&Bs[(f * 16 + l16) * TK];
#pragma unroll
            for (int v = 0; v < 8; ++v)
                bf_[f].u[v] = Brow[(((v & 4) ? 16 : 0) + half * 8 + (v & 3) * 2) >> 1];
        }
#pragma unroll
        for (int f = 0; f < 5; ++f)
            acc[f] = __builtin_amdgcn_wmma_f32_16x16x32_bf16(
                false, af.v, false, bf_[f].v, (short)0, acc[f], false, false);
        __syncthreads();
    }

    // epilogue: D layout — col n = lane&15, row m = half*8 + v
#pragma unroll
    for (int f = 0; f < 5; ++f) {
        int n = n0 + f * 16 + l16;
#pragma unroll
        for (int v = 0; v < 8; ++v) {
            int b = m0 + wave * 16 + half * 8 + v;
            size_t idx = (size_t)b * L + n;
            float a = acc[f][v];
            float d;
            if (MODE == 0)      d = a;
            else if (MODE == 1) d = (a - meanp[b]) * rstdp[b];
            else if (MODE == 2) d = fabsf(a);
            else if (MODE == 3) { float p = aux1[idx]; float aa = fabsf(a);
                                  d = aa < p ? aa : p; }
            else if (MODE == 4) { float t = aux1[idx] - a; float lim = aux2[idx];
                                  float at = fabsf(t); float mn = at < lim ? at : lim;
                                  d = (t >= 0.0f) ? mn : -mn; }
            else if (MODE == 5) d = aux1[idx] - a + aux2[idx];
            else                d = a * aux1[b] + aux2[b];
            D[idx] = d;
        }
    }
}

// ---------------------------------------------------------------------------
// Per-batch stats over xt1 (400): mean/rstd/std + padded normalized xt1p (416).
// ---------------------------------------------------------------------------
__global__ __launch_bounds__(256)
void stats_pad_k(const float* __restrict__ xt1raw, float* __restrict__ meanv,
                 float* __restrict__ rstdv, float* __restrict__ stdv,
                 float* __restrict__ xt1p)
{
    __shared__ float sh[256];
    __shared__ float row[400];
    const int b = blockIdx.x, tid = threadIdx.x;
    float s = 0.0f;
    for (int i = tid; i < 400; i += 256) {
        float v = xt1raw[(size_t)b * 400 + i];
        row[i] = v; s += v;
    }
    sh[tid] = s; __syncthreads();
    for (int o = 128; o > 0; o >>= 1) {
        if (tid < o) sh[tid] += sh[tid + o];
        __syncthreads();
    }
    float mean = sh[0] * (1.0f / 400.0f);
    __syncthreads();
    float s2 = 0.0f;
    for (int i = tid; i < 400; i += 256) {
        float d = row[i] - mean; s2 += d * d;
    }
    sh[tid] = s2; __syncthreads();
    for (int o = 128; o > 0; o >>= 1) {
        if (tid < o) sh[tid] += sh[tid + o];
        __syncthreads();
    }
    float var  = sh[0] * (1.0f / 399.0f) + 1e-8f;   // ddof=1
    float stdd = sqrtf(var);
    float rstd = 1.0f / stdd;
    if (tid == 0) { meanv[b] = mean; rstdv[b] = rstd; stdv[b] = stdd; }
    for (int i = tid; i < 416; i += 256) {
        float v;
        if (i < 8)        v = 2.0f * row[0]   - row[8 - i];
        else if (i < 408) v = row[i - 8];
        else              v = 2.0f * row[399] - row[398 - (i - 408)];
        xt1p[(size_t)b * 416 + i] = (v - mean) * rstd;
    }
}

// ---------------------------------------------------------------------------
// Weight repack: w(Cout,Cin,3) -> packed bf16 pairs Wp[(k>>1)*Npad + n],
// k = t*Cin + ci (low 16 bits = even k). Matches the B-fragment layout.
// ---------------------------------------------------------------------------
__global__ void repack_w_k(const float* __restrict__ w, unsigned int* __restrict__ Wp,
                           int Cin, int Cout, int Npad)
{
    int K = Cin * 3;
    int tot = (K / 2) * Npad;
    int idx = blockIdx.x * 256 + threadIdx.x;
    if (idx >= tot) return;
    int n = idx % Npad, kp = idx / Npad;
    unsigned int out = 0;
    for (int e = 0; e < 2; ++e) {
        int k = kp * 2 + e;
        float v = 0.0f;
        if (n < Cout) {
            int t = k / Cin, ci = k % Cin;
            v = w[((size_t)n * Cin + ci) * 3 + t];
        }
        out |= ((unsigned int)f2bf(v)) << (e * 16);
    }
    Wp[idx] = out;
}

// ---------------------------------------------------------------------------
// conv1: 1 -> 64 ch, K=3 (tiny -> VALU), tanh; position-major output (p,64).
// ---------------------------------------------------------------------------
__global__ void conv1_k(const float* __restrict__ xt1p, const float* __restrict__ w1,
                        const float* __restrict__ b1, float* __restrict__ u1,
                        int b0, int chunkB)
{
    int idx = blockIdx.x * 256 + threadIdx.x;
    int tot = chunkB * 414 * 64;
    if (idx >= tot) return;
    int c = idx & 63;
    int rest = idx >> 6;
    int p = rest % 414;
    int b = rest / 414;
    const float* xp = xt1p + (size_t)(b0 + b) * 416 + p;
    float s = b1[c] + xp[0] * w1[c * 3] + xp[1] * w1[c * 3 + 1] + xp[2] * w1[c * 3 + 2];
    u1[((size_t)b * 414 + p) * 64 + c] = tanhf(s);
}

// ---------------------------------------------------------------------------
// Implicit-GEMM conv: M rows = (batch,out-pos), K=192 (3 taps x 64 in-ch,
// contiguous in position-major layout), N = NFRAG*16 out channels.
// ---------------------------------------------------------------------------
template <int NFRAG, int DO_TANH>
__global__ __launch_bounds__(256)
void conv_gemm_k(const float* __restrict__ Uin, int in_rows,
                 float* __restrict__ Uout, int out_rows,
                 const unsigned int* __restrict__ Wp,
                 const float* __restrict__ bias,
                 int Nreal, int Mtot)
{
    __shared__ unsigned short As[128 * 32];
    const int tid  = threadIdx.x;
    const int wave = tid >> 5;
    const int lane = tid & 31;
    const int half = lane >> 4;
    const int l16  = lane & 15;
    const int m0   = blockIdx.x * 128;
    const int Npad = NFRAG * 16;

    v8f acc[NFRAG];
#pragma unroll
    for (int f = 0; f < NFRAG; ++f)
#pragma unroll
        for (int i = 0; i < 8; ++i) acc[f][i] = 0.0f;

#pragma unroll
    for (int ks = 0; ks < 6; ++ks) {            // K = 192 = 6 * 32
        const int kb = ks * 32;
        for (int i = tid; i < 128 * 8; i += 256) {
            int r = i >> 3, c = i & 7;
            int m = m0 + r;
            float4 v = make_float4(0.f, 0.f, 0.f, 0.f);
            if (m < Mtot) {
                int b = m / out_rows, p = m % out_rows;
                v = *(const float4*)(Uin + ((size_t)b * in_rows + p) * 64 + kb + c * 4);
            }
            unsigned int* d = (unsigned int*)&As[r * 32 + c * 4];
            d[0] = pk2bf(v.x, v.y);
            d[1] = pk2bf(v.z, v.w);
        }
        __syncthreads();

        Frag16 af;
        {
            const unsigned int* Arow =
                (const unsigned int*)&As[(wave * 16 + l16) * 32];
#pragma unroll
            for (int v = 0; v < 8; ++v) {
                int kbase = ((v & 4) ? 16 : 0) + half * 8 + (v & 3) * 2;
                af.u[v] = Arow[kbase >> 1];
            }
        }
        Frag16 bf_[NFRAG];
#pragma unroll
        for (int f = 0; f < NFRAG; ++f)
#pragma unroll
            for (int v = 0; v < 8; ++v)
                bf_[f].u[v] = Wp[(size_t)(ks * 16 + ((v & 4) ? 8 : 0) + half * 4 + (v & 3)) * Npad + f * 16 + l16];
#pragma unroll
        for (int f = 0; f < NFRAG; ++f)
            acc[f] = __builtin_amdgcn_wmma_f32_16x16x32_bf16(
                false, af.v, false, bf_[f].v, (short)0, acc[f], false, false);
        __syncthreads();
    }

#pragma unroll
    for (int f = 0; f < NFRAG; ++f) {
        int n = f * 16 + l16;
#pragma unroll
        for (int v = 0; v < 8; ++v) {
            int m = m0 + wave * 16 + half * 8 + v;
            if (m < Mtot && n < Nreal) {
                float d = acc[f][v] + bias[n];
                if (DO_TANH) d = tanhf(d);
                int b = m / out_rows, p = m % out_rows;
                Uout[((size_t)b * out_rows + p) * Nreal + n] = d;
            }
        }
    }
}

// ---------------------------------------------------------------------------
// Quad-form window weights: w[b,l] = 1 / (|win^T S win| + 0.1).
// ---------------------------------------------------------------------------
__global__ void wquad_k(const float* __restrict__ xt1p, const float* __restrict__ S,
                        float* __restrict__ wbuf, int b0, int chunkB)
{
    __shared__ float Sh[81];
    if (threadIdx.x < 81) Sh[threadIdx.x] = S[threadIdx.x];
    __syncthreads();
    int idx = blockIdx.x * 256 + threadIdx.x;
    int tot = chunkB * 408;
    if (idx >= tot) return;
    int l = idx % 408, b = idx / 408;
    const float* xp = xt1p + (size_t)(b0 + b) * 416 + l;
    float win[9];
#pragma unroll
    for (int k = 0; k < 9; ++k) win[k] = xp[k];
    float a = 0.0f;
#pragma unroll
    for (int k = 0; k < 9; ++k) {
        float s = 0.0f;
#pragma unroll
        for (int m = 0; m < 9; ++m) s += Sh[k * 9 + m] * win[m];
        a += win[k] * s;
    }
    wbuf[(size_t)b * 408 + l] = 1.0f / (fabsf(a) + 0.1f);
}

// ---------------------------------------------------------------------------
// num/den diagonal gather: uflat[b, l*3+n] = sum_j w[l+j]*u4[l+j, n*9+j] /
//                                            sum_j w[l+j]
// ---------------------------------------------------------------------------
__global__ void numden_k(const float* __restrict__ u4c, const float* __restrict__ wbuf,
                         float* __restrict__ uflat, int b0, int chunkB)
{
    int idx = blockIdx.x * 256 + threadIdx.x;
    int tot = chunkB * 400 * 3;
    if (idx >= tot) return;
    int n = idx % 3;
    int rest = idx / 3;
    int l = rest % 400;
    int b = rest / 400;
    float num = 0.0f, den = 0.0f;
#pragma unroll
    for (int j = 0; j < 9; ++j) {
        float wv = wbuf[(size_t)b * 408 + l + j];
        den += wv;
        num += wv * u4c[((size_t)b * 408 + (l + j)) * 27 + n * 9 + j];
    }
    uflat[(size_t)(b0 + b) * 1200 + l * 3 + n] = num / den;
}

// ---------------------------------------------------------------------------
extern "C" void kernel_launch(void* const* d_in, const int* in_sizes, int n_in,
                              void* d_out, int out_size, void* d_ws, size_t ws_size,
                              hipStream_t stream)
{
    (void)in_sizes; (void)n_in; (void)out_size; (void)ws_size;
    const float* xt       = (const float*)d_in[0];
    const float* Average  = (const float*)d_in[1];
    const float* Average1 = (const float*)d_in[2];
    const float* d1       = (const float*)d_in[3];
    const float* d2       = (const float*)d_in[4];
    const float* evl      = (const float*)d_in[5];
    const float* smooth   = (const float*)d_in[6];
    const float* w1       = (const float*)d_in[7];
    const float* b1       = (const float*)d_in[8];
    const float* w2       = (const float*)d_in[9];
    const float* b2       = (const float*)d_in[10];
    const float* w3       = (const float*)d_in[11];
    const float* b3       = (const float*)d_in[12];
    const float* w4       = (const float*)d_in[13];
    const float* b4       = (const float*)d_in[14];
    float* out = (float*)d_out;

    constexpr int Bn = 2048, KF = 1200, LC = 400, CH = 512;
    char* ws = (char*)d_ws;
    size_t off = 0;
    auto alloc = [&](size_t bytes) -> char* {
        char* p = ws + off;
        off += (bytes + 255) & ~(size_t)255;
        return p;
    };
    float* xt1raw = (float*)alloc((size_t)Bn * LC * 4);
    float* meanv  = (float*)alloc((size_t)Bn * 4);
    float* rstdv  = (float*)alloc((size_t)Bn * 4);
    float* stdv   = (float*)alloc((size_t)Bn * 4);
    float* xt1p   = (float*)alloc((size_t)Bn * 416 * 4);
    float* xt11n  = (float*)alloc((size_t)Bn * KF * 4);
    float* minlim = (float*)alloc((size_t)Bn * KF * 4);
    float* uflat  = (float*)alloc((size_t)Bn * KF * 4);
    float* u3cb   = (float*)alloc((size_t)Bn * KF * 4);
    float* u4gb   = (float*)alloc((size_t)Bn * KF * 4);
    unsigned short* Avgb  = (unsigned short*)alloc((size_t)LC * KPAD * 2);
    unsigned short* Avg1b = (unsigned short*)alloc((size_t)KF * KPAD * 2);
    unsigned short* d1b   = (unsigned short*)alloc((size_t)KF * KPAD * 2);
    unsigned short* d2b   = (unsigned short*)alloc((size_t)KF * KPAD * 2);
    unsigned short* evlb  = (unsigned short*)alloc((size_t)KF * KPAD * 2);
    unsigned int* W2p = (unsigned int*)alloc(96 * 64 * 4);
    unsigned int* W3p = (unsigned int*)alloc(96 * 64 * 4);
    unsigned int* W4p = (unsigned int*)alloc(96 * 32 * 4);
    float* u1c  = (float*)alloc((size_t)CH * 414 * 64 * 4);   // also holds u3
    float* u2c  = (float*)alloc((size_t)CH * 412 * 64 * 4);   // also holds u4 (408x27)
    float* wch  = (float*)alloc((size_t)CH * 408 * 4);
    float* u3ch = u1c;
    float* u4ch = u2c;

    dim3 blk(256);

    // pre-convert B-side matrices to zero-padded bf16
    cvt_mat_bf16_k<<<((LC * KPAD) + 255) / 256, blk, 0, stream>>>(Average, Avgb, LC, KF);
    cvt_mat_bf16_k<<<((KF * KPAD) + 255) / 256, blk, 0, stream>>>(Average1, Avg1b, KF, KF);
    cvt_mat_bf16_k<<<((KF * KPAD) + 255) / 256, blk, 0, stream>>>(d1, d1b, KF, KF);
    cvt_mat_bf16_k<<<((KF * KPAD) + 255) / 256, blk, 0, stream>>>(d2, d2b, KF, KF);
    cvt_mat_bf16_k<<<((KF * KPAD) + 255) / 256, blk, 0, stream>>>(evl, evlb, KF, KF);

    // GEMM 1: xt1_raw = xt . Average^T
    gemm_bf16_ep<0, 0><<<dim3(LC / TN, Bn / TM), blk, 0, stream>>>(
        xt, Avgb, xt1raw, KF, LC, nullptr, nullptr, nullptr, nullptr);
    // stats + padded normalized xt1p
    stats_pad_k<<<Bn, blk, 0, stream>>>(xt1raw, meanv, rstdv, stdv, xt1p);
    // GEMM 2: xt11n = normalize(xt . Average1^T)
    gemm_bf16_ep<0, 1><<<dim3(KF / TN, Bn / TM), blk, 0, stream>>>(
        xt, Avg1b, xt11n, KF, KF, meanv, rstdv, nullptr, nullptr);
    // GEMM 3/4: minlimit = min(|d1 . xt_|, |d2 . xt_|)   (xt_ norm fused on A)
    gemm_bf16_ep<1, 2><<<dim3(KF / TN, Bn / TM), blk, 0, stream>>>(
        xt, d1b, minlim, KF, KF, meanv, rstdv, nullptr, nullptr);
    gemm_bf16_ep<1, 3><<<dim3(KF / TN, Bn / TM), blk, 0, stream>>>(
        xt, d2b, minlim, KF, KF, meanv, rstdv, minlim, nullptr);

    // conv weight repacks (bf16 B-fragment layout)
    repack_w_k<<<(96 * 64 + 255) / 256, blk, 0, stream>>>(w2, W2p, 64, 64, 64);
    repack_w_k<<<(96 * 64 + 255) / 256, blk, 0, stream>>>(w3, W3p, 64, 64, 64);
    repack_w_k<<<(96 * 32 + 255) / 256, blk, 0, stream>>>(w4, W4p, 64, 27, 32);

    // conv chain + window weights + num/den, batch-chunked to stay in L2
    for (int b0 = 0; b0 < Bn; b0 += CH) {
        conv1_k<<<(CH * 414 * 64 + 255) / 256, blk, 0, stream>>>(
            xt1p, w1, b1, u1c, b0, CH);
        int M2 = CH * 412;
        conv_gemm_k<4, 1><<<(M2 + 127) / 128, blk, 0, stream>>>(
            u1c, 414, u2c, 412, W2p, b2, 64, M2);
        int M3 = CH * 410;
        conv_gemm_k<4, 1><<<(M3 + 127) / 128, blk, 0, stream>>>(
            u2c, 412, u3ch, 410, W3p, b3, 64, M3);
        int M4 = CH * 408;
        conv_gemm_k<2, 0><<<(M4 + 127) / 128, blk, 0, stream>>>(
            u3ch, 410, u4ch, 408, W4p, b4, 27, M4);
        wquad_k<<<(CH * 408 + 255) / 256, blk, 0, stream>>>(
            xt1p, smooth, wch, b0, CH);
        numden_k<<<(CH * 400 * 3 + 255) / 256, blk, 0, stream>>>(
            u4ch, wch, uflat, b0, CH);
    }

    // GEMM 5: u3cb = clamp(uflat - Average1.uflat, minlim)
    gemm_bf16_ep<0, 4><<<dim3(KF / TN, Bn / TM), blk, 0, stream>>>(
        uflat, Avg1b, u3cb, KF, KF, nullptr, nullptr, uflat, minlim);
    // GEMM 6: u4gb = u3cb - Average1.u3cb + xt11n
    gemm_bf16_ep<0, 5><<<dim3(KF / TN, Bn / TM), blk, 0, stream>>>(
        u3cb, Avg1b, u4gb, KF, KF, nullptr, nullptr, u3cb, xt11n);
    // GEMM 7: out = (evl . u4gb) * std + mean
    gemm_bf16_ep<0, 6><<<dim3(KF / TN, Bn / TM), blk, 0, stream>>>(
        u4gb, evlb, out, KF, KF, nullptr, nullptr, stdv, meanv);
}
